// DeepseekV2DecoderLayer_61220463837456
// MI455X (gfx1250) — matrix operand order repaired and checked by reading, not compiled, not run on hardware
//
#include <hip/hip_runtime.h>
#include <hip/hip_bf16.h>

// ---------------- CDNA5 WMMA types ----------------
typedef __attribute__((ext_vector_type(16))) __bf16 v16bf;
typedef __attribute__((ext_vector_type(8)))  float  v8f;

// ---------------- constants ----------------
#define SEQ    2048
#define HIDDEN 2048
#define NHEAD  16
#define NOPE   128
#define ROPE   64
#define VDIM   128
#define QLR    1536
#define KVLR   512
#define QK     192      // NOPE + ROPE
#define QKVA_N 2112     // QLR + KVLR + ROPE

// ---------------- helpers ----------------
__device__ __forceinline__ unsigned short f2bf(float f) {
    unsigned int u = __float_as_uint(f);
    unsigned int r = u + 0x7FFFu + ((u >> 16) & 1u);   // round-to-nearest-even
    return (unsigned short)(r >> 16);
}
__device__ __forceinline__ float bf2f(unsigned short s) {
    return __uint_as_float(((unsigned int)s) << 16);
}
__device__ __forceinline__ unsigned int pack2bf(float a, float b) {
    return (unsigned int)f2bf(a) | ((unsigned int)f2bf(b) << 16);
}

// ---- CDNA5 async global->LDS copy (ASYNCcnt-tracked, bypasses VGPRs) ----
// LDS address is taken from the vdst VGPR; low 32 bits of a generic shared pointer are
// the LDS byte offset, so ptrtoint + truncate gives the vdst operand.
__device__ __forceinline__ void async_copy16(const void* gsrc, void* lds_dst) {
    unsigned loff = (unsigned)(unsigned long long)lds_dst;
    unsigned long long ga = (unsigned long long)gsrc;
    asm volatile("global_load_async_to_lds_b128 %0, %1, off"
                 :: "v"(loff), "v"(ga) : "memory");
}
__device__ __forceinline__ void async_wait0() {
    asm volatile("s_wait_asynccnt 0x0" ::: "memory");
}

// A-matrix 16x32 bf16 fragment (ISA 7.12.2): lane holds row M=lane%16,
// VGPR v holds K pair at (v/4)*16 + (lane/16)*8 + (v%4)*2.
template<int LD>
__device__ __forceinline__ v16bf load_frag_a(const unsigned short* base, int row, int kk, int lane) {
    union { v16bf v; unsigned int u[8]; } f;
    const unsigned int* p = (const unsigned int*)(base + row * LD + kk * 32);
    int half8 = (lane >> 4) << 3;
#pragma unroll
    for (int r = 0; r < 8; ++r) {
        int kb = ((r >> 2) << 4) + half8 + ((r & 3) << 1);
        f.u[r] = p[kb >> 1];
    }
    return f.v;
}
// B-matrix 32x16 bf16 fragment: lane holds col N=lane%16,
// VGPR v holds K pair at (v/4)*16 + (lane/16)*8 + (v%4)*2 (same as A).
template<int LD>
__device__ __forceinline__ v16bf load_frag_b(const unsigned short* base, int row, int kk, int lane) {
    union { v16bf v; unsigned int u[8]; } f;
    const unsigned int* p = (const unsigned int*)(base + row * LD + kk * 32);
    int koff = (lane >> 4) << 3;
#pragma unroll
    for (int r = 0; r < 8; ++r) {
        int kb = ((r >> 2) << 4) + koff + ((r & 3) << 1);
        f.u[r] = p[kb >> 1];
    }
    return f.v;
}

// ---------------- generic GEMM: C[M,N] = A[M,K] @ B[N,K]^T ----------------
// A: f32 or bf16 (template), B: f32 weights, C: f32 or bf16 (template).
// BM=BN=64, BK=64, 256 threads = 8 wave32 waves in a 4(row)x2(col) layout;
// each wave runs 4 WMMAs per load phase into two 16x16 f32 accumulators.
// bf16 A-tiles are moved with global_load_async_to_lds_b128.
template<bool A_BF16, bool C_BF16>
__global__ __launch_bounds__(256) void gemm_bt_kernel(
    const void* __restrict__ Ap, const float* __restrict__ B, void* __restrict__ Cp,
    int M, int N, int K, int lda, int ldb, int ldc) {
    __shared__ __align__(16) unsigned short As[64 * 64];   // 8 KB
    __shared__ __align__(16) unsigned short Bs[64 * 64];   // 8 KB

    const int n0 = blockIdx.x * 64;
    const int m0 = blockIdx.y * 64;
    const int t = threadIdx.x;
    const int lane = t & 31, w = t >> 5;
    const int wr = w >> 1, wc = w & 1;
    const int half = lane >> 4, ln = lane & 15;
    const int r = t >> 2, c0 = (t & 3) * 16;   // tile-load coords: 16 elems/thread

    v8f acc[2];
#pragma unroll
    for (int s = 0; s < 2; ++s)
#pragma unroll
        for (int v = 0; v < 8; ++v) acc[s][v] = 0.0f;

    for (int k0 = 0; k0 < K; k0 += 64) {
        if (A_BF16) {
            const unsigned short* A = (const unsigned short*)Ap;
            const unsigned short* src = &A[(size_t)(m0 + r) * lda + k0 + c0];
            async_copy16(src,     &As[r * 64 + c0]);
            async_copy16(src + 8, &As[r * 64 + c0 + 8]);
        } else {
            const float* A = (const float*)Ap;
            const float* src = &A[(size_t)(m0 + r) * lda + k0 + c0];
            unsigned int* dst = (unsigned int*)&As[r * 64 + c0];
#pragma unroll
            for (int q = 0; q < 4; ++q) {
                float4 f = *(const float4*)(src + 4 * q);
                dst[2 * q]     = pack2bf(f.x, f.y);
                dst[2 * q + 1] = pack2bf(f.z, f.w);
            }
        }
        {
            const float* src = &B[(size_t)(n0 + r) * ldb + k0 + c0];
            unsigned int* dst = (unsigned int*)&Bs[r * 64 + c0];
#pragma unroll
            for (int q = 0; q < 4; ++q) {
                float4 f = *(const float4*)(src + 4 * q);
                dst[2 * q]     = pack2bf(f.x, f.y);
                dst[2 * q + 1] = pack2bf(f.z, f.w);
            }
        }
        if (k0 + 64 < K)   // pull next B k-tile toward L2 while WMMAs run
            __builtin_prefetch(&B[(size_t)(n0 + r) * ldb + k0 + 64 + c0], 0, 1);
        if (A_BF16) async_wait0();
        __syncthreads();

#pragma unroll
        for (int kk = 0; kk < 2; ++kk) {
            v16bf a = load_frag_a<64>(As, 16 * wr + ln, kk, lane);
#pragma unroll
            for (int s = 0; s < 2; ++s) {
                v16bf b = load_frag_b<64>(Bs, 32 * wc + 16 * s + ln, kk, lane);
                acc[s] = __builtin_amdgcn_wmma_f32_16x16x32_bf16(
                    false, a, false, b, (short)0, acc[s], false, false);
            }
        }
        __syncthreads();
    }

#pragma unroll
    for (int v = 0; v < 8; ++v) {
        int m = m0 + 16 * wr + v + 8 * half;
#pragma unroll
        for (int s = 0; s < 2; ++s) {
            int n = n0 + 32 * wc + 16 * s + ln;
            if (C_BF16)
                ((unsigned short*)Cp)[(size_t)m * ldc + n] = f2bf(acc[s][v]);
            else
                ((float*)Cp)[(size_t)m * ldc + n] = acc[s][v];
        }
    }
}

// ---------------- RMSNorm (f32 in -> bf16 out) ----------------
__global__ __launch_bounds__(256) void rmsnorm_kernel(
    const float* __restrict__ in, int ld, int col0, int width,
    const float* __restrict__ w, unsigned short* __restrict__ out, int ldo) {
    __shared__ float red[256];
    const int row = blockIdx.x;
    const float* x = in + (size_t)row * ld + col0;
    float s = 0.0f;
    for (int c = threadIdx.x; c < width; c += 256) { float v = x[c]; s += v * v; }
    red[threadIdx.x] = s;
    __syncthreads();
    for (int k = 128; k > 0; k >>= 1) {
        if (threadIdx.x < k) red[threadIdx.x] += red[threadIdx.x + k];
        __syncthreads();
    }
    float inv = rsqrtf(red[0] / (float)width + 1e-6f);
    for (int c = threadIdx.x; c < width; c += 256)
        out[(size_t)row * ldo + c] = f2bf(x[c] * inv * w[c]);
}

// ---------------- RoPE + repack into per-head attention layouts ----------------
// Qh[h][s][192], Kh[h][s][192] (k_pe broadcast to all heads), VT[h][d][s].
__global__ __launch_bounds__(256) void pack_rope_kernel(
    const float* __restrict__ qkv_a,            // [S][2112] f32 (k_pe source)
    const unsigned short* __restrict__ qb,      // [S][3072] bf16
    const unsigned short* __restrict__ kvb,     // [S][4096] bf16
    const int* __restrict__ positions,
    unsigned short* __restrict__ Qh, unsigned short* __restrict__ Kh,
    unsigned short* __restrict__ VT) {
    const int s = blockIdx.x;
    const int t = threadIdx.x;
    const float pos = (float)positions[s];
    __shared__ unsigned short kpe[ROPE];

    if (t < ROPE / 2) {
        float x1 = qkv_a[(size_t)s * QKVA_N + QLR + KVLR + 2 * t];
        float x2 = qkv_a[(size_t)s * QKVA_N + QLR + KVLR + 2 * t + 1];
        float inv_freq = __powf(10000.0f, -((float)(2 * t) / (float)ROPE));
        float f = pos * inv_freq;
        float c = __cosf(f), sn = __sinf(f);
        kpe[2 * t]     = f2bf(x1 * c - x2 * sn);
        kpe[2 * t + 1] = f2bf(x2 * c + x1 * sn);
    }
    __syncthreads();

    // Q: nope passthrough, pe gets RoPE
    for (int i = t; i < NHEAD * QK; i += 256) {
        int h = i / QK, d = i % QK;
        unsigned short val;
        if (d < NOPE) {
            val = qb[(size_t)s * (NHEAD * QK) + h * QK + d];
        } else {
            int dd = d - NOPE, p = dd >> 1;
            float x1 = bf2f(qb[(size_t)s * (NHEAD * QK) + h * QK + NOPE + 2 * p]);
            float x2 = bf2f(qb[(size_t)s * (NHEAD * QK) + h * QK + NOPE + 2 * p + 1]);
            float inv_freq = __powf(10000.0f, -((float)(2 * p) / (float)ROPE));
            float f = pos * inv_freq;
            float c = __cosf(f), sn = __sinf(f);
            val = (dd & 1) ? f2bf(x2 * c + x1 * sn) : f2bf(x1 * c - x2 * sn);
        }
        Qh[((size_t)h * SEQ + s) * QK + d] = val;
    }
    // K: k_nope per head + broadcast roped k_pe
    for (int i = t; i < NHEAD * QK; i += 256) {
        int h = i / QK, d = i % QK;
        unsigned short val = (d < NOPE)
            ? kvb[(size_t)s * (NHEAD * (NOPE + VDIM)) + h * (NOPE + VDIM) + d]
            : kpe[d - NOPE];
        Kh[((size_t)h * SEQ + s) * QK + d] = val;
    }
    // V transposed: VT[h][d][s]
    for (int i = t; i < NHEAD * VDIM; i += 256) {
        int h = i / VDIM, d = i % VDIM;
        VT[((size_t)h * VDIM + d) * SEQ + s] =
            kvb[(size_t)s * (NHEAD * (NOPE + VDIM)) + h * (NOPE + VDIM) + NOPE + d];
    }
}

// ---------------- Flash attention (per head, 64 query rows / block) ----------------
__global__ __launch_bounds__(256) void mla_attn_kernel(
    const unsigned short* __restrict__ Qh, const unsigned short* __restrict__ Kh,
    const unsigned short* __restrict__ VT, float* __restrict__ Oout) {
    __shared__ __align__(16) unsigned short Qs[64 * QK];     // 24 KB
    __shared__ __align__(16) unsigned short Ks[64 * QK];     // 24 KB
    __shared__ __align__(16) unsigned short VTs[VDIM * 64];  // 16 KB
    __shared__ __align__(16) float          Stile[64 * 64];  // 16 KB
    __shared__ __align__(16) unsigned short Pt[64 * 64];     // 8 KB
    __shared__ float mstate[64], lstate[64], sfac[64];

    const int bx = blockIdx.x, h = blockIdx.y;
    const int qm = bx * 64;
    const int t = threadIdx.x;
    const int lane = t & 31, w = t >> 5;
    const int wr = w >> 1, wc = w & 1;
    const int half = lane >> 4, ln = lane & 15;
    const float scale = 0.0721687836487032f; // 1/sqrt(192)

    // async-stage Q block (waited inside the first loop iteration)
    for (int i = t; i < 64 * QK / 8; i += 256) {
        int e = i * 8, sl = e / QK, d = e % QK;
        async_copy16(&Qh[((size_t)h * SEQ + qm + sl) * QK + d], &Qs[e]);
    }
    if (t < 64) { mstate[t] = -3.0e38f; lstate[t] = 0.0f; }

    v8f oacc[4];
#pragma unroll
    for (int s = 0; s < 4; ++s)
#pragma unroll
        for (int v = 0; v < 8; ++v) oacc[s][v] = 0.0f;

    for (int kb = 0; kb <= bx; ++kb) {
        __syncthreads();  // previous-iteration consumers done before overwrite
        for (int i = t; i < 64 * QK / 8; i += 256) {
            int e = i * 8, sl = e / QK, d = e % QK;
            async_copy16(&Kh[((size_t)h * SEQ + kb * 64 + sl) * QK + d], &Ks[e]);
        }
        for (int i = t; i < VDIM * 64 / 8; i += 256) {
            int e = i * 8, d = e / 64, n = e % 64;
            async_copy16(&VT[((size_t)h * VDIM + d) * SEQ + kb * 64 + n], &VTs[e]);
        }
        if (kb < bx)  // warm L2 with the next K block while we compute
            __builtin_prefetch(&Kh[((size_t)h * SEQ + (kb + 1) * 64) * QK + t * 8], 0, 1);
        async_wait0();
        __syncthreads();

        // S-tile = Q(64x192) @ K(64x192)^T
        v8f sacc[2];
#pragma unroll
        for (int s = 0; s < 2; ++s)
#pragma unroll
            for (int v = 0; v < 8; ++v) sacc[s][v] = 0.0f;
#pragma unroll
        for (int kk = 0; kk < QK / 32; ++kk) {
            v16bf a = load_frag_a<QK>(Qs, 16 * wr + ln, kk, lane);
#pragma unroll
            for (int s = 0; s < 2; ++s) {
                v16bf b = load_frag_b<QK>(Ks, 32 * wc + 16 * s + ln, kk, lane);
                sacc[s] = __builtin_amdgcn_wmma_f32_16x16x32_bf16(
                    false, a, false, b, (short)0, sacc[s], false, false);
            }
        }
        // scale + causal mask -> LDS
#pragma unroll
        for (int s = 0; s < 2; ++s)
#pragma unroll
            for (int v = 0; v < 8; ++v) {
                int m = 16 * wr + v + 8 * half;
                int n = 32 * wc + 16 * s + ln;
                float val = sacc[s][v] * scale;
                if (kb == bx && (kb * 64 + n) > (qm + m)) val = -3.0e38f;
                Stile[m * 64 + n] = val;
            }
        __syncthreads();

        // online softmax per row
        if (t < 64) {
            int rr = t;
            float mp = mstate[rr];
            float rowmax = -3.0e38f;
            for (int c = 0; c < 64; ++c) rowmax = fmaxf(rowmax, Stile[rr * 64 + c]);
            float mn = fmaxf(mp, rowmax);
            float fsc = __expf(mp - mn);
            float rs = 0.0f;
            for (int c = 0; c < 64; ++c) {
                float p = __expf(Stile[rr * 64 + c] - mn);
                Pt[rr * 64 + c] = f2bf(p);
                rs += p;
            }
            mstate[rr] = mn;
            lstate[rr] = lstate[rr] * fsc + rs;
            sfac[rr] = fsc;
        }
        __syncthreads();

        // rescale running output accumulators
#pragma unroll
        for (int v = 0; v < 8; ++v) {
            int m = 16 * wr + v + 8 * half;
            float f = sfac[m];
#pragma unroll
            for (int s = 0; s < 4; ++s) oacc[s][v] *= f;
        }
        // O += P(64x64) @ V^T(128x64)^T   (A@B^T form via transposed V)
#pragma unroll
        for (int kk = 0; kk < 2; ++kk) {
            v16bf a = load_frag_a<64>(Pt, 16 * wr + ln, kk, lane);
#pragma unroll
            for (int s = 0; s < 4; ++s) {
                v16bf b = load_frag_b<64>(VTs, 64 * wc + 16 * s + ln, kk, lane);
                oacc[s] = __builtin_amdgcn_wmma_f32_16x16x32_bf16(
                    false, a, false, b, (short)0, oacc[s], false, false);
            }
        }
    }
    __syncthreads();

#pragma unroll
    for (int v = 0; v < 8; ++v) {
        int m = 16 * wr + v + 8 * half;
        float invl = 1.0f / lstate[m];
#pragma unroll
        for (int s = 0; s < 4; ++s) {
            int n = 64 * wc + 16 * s + ln;
            Oout[(size_t)(qm + m) * (NHEAD * VDIM) + h * VDIM + n] = oacc[s][v] * invl;
        }
    }
}

// ---------------- launch ----------------
extern "C" void kernel_launch(void* const* d_in, const int* in_sizes, int n_in,
                              void* d_out, int out_size, void* d_ws, size_t ws_size,
                              hipStream_t stream) {
    const float* hidden    = (const float*)d_in[0];   // [2048][2048]
    const float* w_qkv_a   = (const float*)d_in[1];   // [2112][2048]
    const float* q_a_ln_w  = (const float*)d_in[2];   // [1536]
    const float* w_q_b     = (const float*)d_in[3];   // [3072][1536]
    const float* w_kv_b    = (const float*)d_in[4];   // [4096][512]
    const float* kv_a_ln_w = (const float*)d_in[5];   // [512]
    const float* w_o       = (const float*)d_in[6];   // [2048][2048]
    const int*   positions = (const int*)d_in[7];     // [2048]
    float* out = (float*)d_out;

    char* p = (char*)d_ws;
    float*          qkv_a  = (float*)p;          p += (size_t)SEQ * QKVA_N * 4;
    unsigned short* q_a_n  = (unsigned short*)p; p += (size_t)SEQ * QLR * 2;
    unsigned short* kv_a_n = (unsigned short*)p; p += (size_t)SEQ * KVLR * 2;
    unsigned short* q_bf   = (unsigned short*)p; p += (size_t)SEQ * NHEAD * QK * 2;
    unsigned short* kv_bf  = (unsigned short*)p; p += (size_t)SEQ * NHEAD * (NOPE + VDIM) * 2;
    unsigned short* Qhp    = (unsigned short*)p; p += (size_t)NHEAD * SEQ * QK * 2;
    unsigned short* Khp    = (unsigned short*)p; p += (size_t)NHEAD * SEQ * QK * 2;
    unsigned short* VTp    = (unsigned short*)p; p += (size_t)NHEAD * VDIM * SEQ * 2;
    float*          attn_o = (float*)p;          p += (size_t)SEQ * NHEAD * VDIM * 4;

    dim3 blk(256);

    // 1) qkv_a = hidden @ w_qkv_a^T   [2048 x 2112]
    gemm_bt_kernel<false, false><<<dim3(QKVA_N / 64, SEQ / 64), blk, 0, stream>>>(
        hidden, w_qkv_a, qkv_a, SEQ, QKVA_N, HIDDEN, HIDDEN, HIDDEN, QKVA_N);
    // 2-3) RMSNorm branches -> bf16
    rmsnorm_kernel<<<SEQ, blk, 0, stream>>>(qkv_a, QKVA_N, 0, QLR, q_a_ln_w, q_a_n, QLR);
    rmsnorm_kernel<<<SEQ, blk, 0, stream>>>(qkv_a, QKVA_N, QLR, KVLR, kv_a_ln_w, kv_a_n, KVLR);
    // 4) q = q_a_n @ w_q_b^T   [2048 x 3072] bf16
    gemm_bt_kernel<true, true><<<dim3((NHEAD * QK) / 64, SEQ / 64), blk, 0, stream>>>(
        q_a_n, w_q_b, q_bf, SEQ, NHEAD * QK, QLR, QLR, QLR, NHEAD * QK);
    // 5) kv = kv_a_n @ w_kv_b^T  [2048 x 4096] bf16
    gemm_bt_kernel<true, true><<<dim3((NHEAD * (NOPE + VDIM)) / 64, SEQ / 64), blk, 0, stream>>>(
        kv_a_n, w_kv_b, kv_bf, SEQ, NHEAD * (NOPE + VDIM), KVLR, KVLR, KVLR, NHEAD * (NOPE + VDIM));
    // 6) RoPE + repack per-head Q/K/V^T
    pack_rope_kernel<<<SEQ, blk, 0, stream>>>(qkv_a, q_bf, kv_bf, positions, Qhp, Khp, VTp);
    // 7) causal flash attention
    mla_attn_kernel<<<dim3(SEQ / 64, NHEAD), blk, 0, stream>>>(Qhp, Khp, VTp, attn_o);
    // 8) out = attn_o @ w_o^T   [2048 x 2048] f32
    gemm_bt_kernel<false, false><<<dim3(HIDDEN / 64, SEQ / 64), blk, 0, stream>>>(
        attn_o, w_o, out, SEQ, HIDDEN, NHEAD * VDIM, NHEAD * VDIM, NHEAD * VDIM, HIDDEN);
}
